// AcidBaseDense_73375221285497
// MI455X (gfx1250) — matrix operand order repaired and checked by reading, not compiled, not run on hardware
//
#include <hip/hip_runtime.h>
#include <stdint.h>

typedef __attribute__((ext_vector_type(16))) _Float16 v16h;
typedef __attribute__((ext_vector_type(8)))  _Float16 v8h;
typedef __attribute__((ext_vector_type(4)))  _Float16 v4h;
typedef __attribute__((ext_vector_type(8)))  float    v8f;

#define DIM   4096
#define BM    128
#define BN    128
#define BK    32
#define LDA   40                 // halves per LDS row: 32 + 8 pad -> 80B stride (16B aligned)
#define NKT   (DIM / BK)         // 128 K-tiles
#define ABYTES (BM * LDA * 2)    // one LDS A buffer: 10240 B
#define BBYTES (BN * LDA * 2)

#define LOG2_10      3.3219280948873623f
#define INV_LOG2_10  0.30102999566398120f

__device__ __forceinline__ v8f zero_v8f() {
    v8f z;
    #pragma unroll
    for (int i = 0; i < 8; ++i) z[i] = 0.0f;
    return z;
}

// ---------------- prep 1: x -> a = 0.1*(10^-x - 1e-14*10^x), f16 [M][K] ----------------
__global__ __launch_bounds__(256)
void prep_x_f16(const float* __restrict__ X, _Float16* __restrict__ Ah)
{
    size_t i = (size_t)blockIdx.x * 256 + threadIdx.x;     // over DIM*DIM/4 float4s
    float4 v = reinterpret_cast<const float4*>(X)[i];
    float f[4] = {v.x, v.y, v.z, v.w};
    v4h h;
    #pragma unroll
    for (int u = 0; u < 4; ++u) {
        float e = __builtin_amdgcn_exp2f(-f[u] * LOG2_10);
        float a = 0.1f * (e - 1.0e-14f * __builtin_amdgcn_rcpf(e));
        h[u] = (_Float16)a;
    }
    reinterpret_cast<v4h*>(Ah)[i] = h;
}

// ---------------- prep 2: w [K][N] -> Bt = sign(w)^T, f16 [N][K] (LDS transpose) ------
__global__ __launch_bounds__(256)
void prep_w_sign_t(const float* __restrict__ W, _Float16* __restrict__ Bt)
{
    __shared__ _Float16 t[64][72];            // 64x64 tile, 72-half row pad (144B, 16B-mult)
    const int tid = threadIdx.x;
    const int n0  = blockIdx.x * 64;
    const int k0  = blockIdx.y * 64;

    #pragma unroll
    for (int it = 0; it < 4; ++it) {
        int idx = tid + it * 256;             // 0..1023 : 64 rows x 16 float4
        int r   = idx >> 4;                   // k within tile
        int c4  = idx & 15;                   // float4 along N
        float4 v = *reinterpret_cast<const float4*>(W + (size_t)(k0 + r) * DIM + n0 + c4 * 4);
        float f[4] = {v.x, v.y, v.z, v.w};
        #pragma unroll
        for (int u = 0; u < 4; ++u) {
            float s = (f[u] > 0.0f) ? 1.0f : ((f[u] < 0.0f) ? -1.0f : 0.0f);
            t[c4 * 4 + u][r] = (_Float16)s;   // transposed store
        }
    }
    __syncthreads();
    #pragma unroll
    for (int it = 0; it < 2; ++it) {
        int idx = tid + it * 256;             // 0..511 : 64 rows x 8 chunks of 8 halves
        int n   = idx >> 3;
        int ch  = idx & 7;
        v8h val = *reinterpret_cast<const v8h*>(&t[n][ch * 8]);
        *reinterpret_cast<v8h*>(Bt + (size_t)(n0 + n) * DIM + k0 + ch * 8) = val;
    }
}

// ---------------- GEMM: async LDS DMA + WMMA f16, fused pH epilogue -------------------
__global__ __launch_bounds__(256)
void acidbase_wmma_async(const _Float16* __restrict__ Ah,
                         const _Float16* __restrict__ Bt,
                         float* __restrict__ OUT)
{
    __shared__ _Float16 As[2][BM * LDA];      // [M][K] tiles (padded rows)
    __shared__ _Float16 Bs[2][BN * LDA];      // [N][K] tiles (padded rows)

    const int tid   = threadIdx.x;
    const int lane  = tid & 31;
    const int wv    = tid >> 5;               // 8 waves
    const int wm    = wv & 1;                 // 2 waves along M (64 rows)
    const int wn    = wv >> 1;                // 4 waves along N (32 cols)
    const int lhalf = lane & 15;
    const int sel   = lane >> 4;

    const int m0 = blockIdx.y * BM;
    const int n0 = blockIdx.x * BN;

    // LDS byte offsets (flat-address low 32 bits == LDS offset, ISA 10.2)
    const uint32_t as_base = (uint32_t)(size_t)&As[0][0];
    const uint32_t bs_base = (uint32_t)(size_t)&Bs[0][0];
    const uint64_t a_g = (uint64_t)Ah;
    const uint64_t b_g = (uint64_t)Bt;

    // Issue ASYNCcnt-tracked DMA: global (f16 row-major) -> padded LDS tile.
    // Each wave instr moves 32 lanes x 16B = 512B (8 rows, 4 chunks/row).
    auto issue_async = [&](int kt, int buf) {
        const uint32_t koff = (uint32_t)kt * (BK * 2);          // byte offset along K
        const int c16 = lane & 3;
        #pragma unroll
        for (int i = 0; i < 2; ++i) {
            const int row = wv * 16 + i * 8 + (lane >> 2);      // 0..127 across 8 waves
            uint32_t ga = (uint32_t)(m0 + row) * (DIM * 2) + koff + c16 * 16;
            uint32_t la = as_base + (uint32_t)buf * ABYTES + (uint32_t)row * (LDA * 2) + c16 * 16;
            asm volatile("global_load_async_to_lds_b128 %0, %1, %2"
                         :: "v"(la), "v"(ga), "s"(a_g) : "memory");
            uint32_t gb = (uint32_t)(n0 + row) * (DIM * 2) + koff + c16 * 16;
            uint32_t lb = bs_base + (uint32_t)buf * BBYTES + (uint32_t)row * (LDA * 2) + c16 * 16;
            asm volatile("global_load_async_to_lds_b128 %0, %1, %2"
                         :: "v"(lb), "v"(gb), "s"(b_g) : "memory");
        }
    };

    // WMMA fragment loads from padded LDS (ISA 7.12.2 16-bit layouts)
    auto ldA = [&](int buf, int i) -> v16h {
        const _Float16* p = &As[buf][(wm * 64 + i * 16 + lhalf) * LDA];
        v8h lo = *reinterpret_cast<const v8h*>(p + sel * 8);
        v8h hi = *reinterpret_cast<const v8h*>(p + 16 + sel * 8);
        return __builtin_shufflevector(lo, hi, 0,1,2,3,4,5,6,7,8,9,10,11,12,13,14,15);
    };
    auto ldB = [&](int buf, int j) -> v16h {
        const _Float16* p = &Bs[buf][(wn * 32 + j * 16 + lhalf) * LDA];
        v8h lo = *reinterpret_cast<const v8h*>(p + sel * 8);
        v8h hi = *reinterpret_cast<const v8h*>(p + 16 + sel * 8);
        return __builtin_shufflevector(lo, hi, 0,1,2,3,4,5,6,7,8,9,10,11,12,13,14,15);
    };

    v8f acc[4][2];
    #pragma unroll
    for (int i = 0; i < 4; ++i)
        #pragma unroll
        for (int j = 0; j < 2; ++j)
            acc[i][j] = zero_v8f();

    issue_async(0, 0);
    asm volatile("s_wait_asynccnt 0x0" ::: "memory");
    __syncthreads();

    for (int kt = 0; kt < NKT; ++kt) {
        const int buf = kt & 1;
        if (kt + 1 < NKT) issue_async(kt + 1, buf ^ 1);   // DMA overlaps WMMA below

        v16h bfr[2];
        #pragma unroll
        for (int j = 0; j < 2; ++j) bfr[j] = ldB(buf, j);
        #pragma unroll
        for (int i = 0; i < 4; ++i) {
            v16h af = ldA(buf, i);
            #pragma unroll
            for (int j = 0; j < 2; ++j)
                acc[i][j] = __builtin_amdgcn_wmma_f32_16x16x32_f16(
                    false, af, false, bfr[j], (short)0, acc[i][j], false, false);
        }

        if (kt + 1 < NKT) {
            asm volatile("s_wait_asynccnt 0x0" ::: "memory");
            __syncthreads();
        }
    }

    // ---- epilogue: titration pH ----
    // C/D layout: VGPR v, lanes 0-15 -> M=v, lanes 16-31 -> M=8+v; N=lane&15
    #pragma unroll
    for (int i = 0; i < 4; ++i) {
        #pragma unroll
        for (int j = 0; j < 2; ++j) {
            const int col = n0 + wn * 32 + j * 16 + lhalf;
            #pragma unroll
            for (int v = 0; v < 8; ++v) {
                const int row = m0 + wm * 64 + i * 16 + sel * 8 + v;
                float r    = acc[i][j][v];
                float conc = fabsf(r) * (1.0f / 409.6f);  // / (n_in * transfer_volume)
                float l10  = __builtin_amdgcn_logf(conc) * INV_LOG2_10;
                float ph   = (r < 0.0f) ? (14.0f + l10) : (-l10);
                OUT[(size_t)row * DIM + col] = ph;
            }
        }
    }
}

extern "C" void kernel_launch(void* const* d_in, const int* in_sizes, int n_in,
                              void* d_out, int out_size, void* d_ws, size_t ws_size,
                              hipStream_t stream) {
    (void)in_sizes; (void)n_in; (void)out_size; (void)ws_size;
    const float* x = (const float*)d_in[0];
    const float* w = (const float*)d_in[1];
    float* out = (float*)d_out;

    _Float16* Ah = (_Float16*)d_ws;                                   // 32 MB
    _Float16* Bt = (_Float16*)((char*)d_ws + (size_t)DIM * DIM * 2);  // 32 MB

    hipLaunchKernelGGL(prep_x_f16, dim3(DIM * (DIM / 4) / 256), dim3(256), 0, stream, x, Ah);
    hipLaunchKernelGGL(prep_w_sign_t, dim3(DIM / 64, DIM / 64), dim3(256), 0, stream, w, Bt);
    hipLaunchKernelGGL(acidbase_wmma_async, dim3(DIM / BN, DIM / BM), dim3(256), 0, stream,
                       Ah, Bt, out);
}
